// MultiHeadAttention_33921651704164
// MI455X (gfx1250) — matrix operand order repaired and checked by reading, not compiled, not run on hardware
//
#include <hip/hip_runtime.h>
#include <hip/hip_bf16.h>

typedef __attribute__((ext_vector_type(16))) __bf16 v16bf;
typedef __attribute__((ext_vector_type(8)))  __bf16 v8bf;
typedef __attribute__((ext_vector_type(8)))  float  v8f;

union BF16x16 { v16bf v; v8bf h[2]; };

constexpr int DM    = 1024;  // d_model (= N = K of projections)
constexpr int S_LEN = 2048;
constexpr int BATCH = 4;
constexpr int HEADS = 16;
constexpr int DH    = 64;    // head dim

__device__ __forceinline__ v8f wmma_bf16(v16bf a, v16bf b, v8f c) {
    // (neg_a, A, neg_b, B, c_mod, C, reuse_a, reuse_b) -> v_wmma_f32_16x16x32_bf16
    return __builtin_amdgcn_wmma_f32_16x16x32_bf16(false, a, false, b, (short)0, c,
                                                   false, false);
}

// ---------------------------------------------------------------------------
// GEMM: Out[M,N] = A[M,K] * W[K,N] + bias, K = N = 1024, M = B*S = 8192.
// A_BF16: A is bf16 (context) vs f32 (query).
// OUT_SPLIT: write bf16 in [B,H,S,DH] head-split layout vs f32 flat [M,N].
// Block tile 128x64, 8 waves, each wave 16 rows x 64 cols (4 C tiles).
// ---------------------------------------------------------------------------
template <bool A_BF16, bool OUT_SPLIT>
__global__ __launch_bounds__(256) void gemm_bias_kernel(
    const void* __restrict__ Ap, const float* __restrict__ W,
    const float* __restrict__ bias, void* __restrict__ Out) {
    constexpr int BM = 128, BN = 64, BK = 32;
    constexpr int LDA = 40;  // padded LDS row stride (elements)
    constexpr int LDB = 40;
    __shared__ __bf16 lA[BM * LDA];   // A tile, row-major [row][k]
    __shared__ __bf16 lBt[BN * LDB];  // B tile transposed [n][k]

    const int tid  = threadIdx.x;
    const int wv   = tid >> 5;
    const int lane = tid & 31;
    const int lrow = lane & 15;
    const int hf   = lane >> 4;
    const int gm   = blockIdx.y * BM;
    const int gn   = blockIdx.x * BN;

    v8f acc[4] = {};

    for (int k0 = 0; k0 < DM; k0 += BK) {
        __syncthreads();
        {   // A tile: 128x32, thread loads 16 contiguous elements
            const int r  = tid >> 1;
            const int cb = (tid & 1) * 16;
            if (A_BF16) {
                const __bf16* src = (const __bf16*)Ap + (size_t)(gm + r) * DM + k0 + cb;
                #pragma unroll
                for (int i = 0; i < 16; ++i) lA[r * LDA + cb + i] = src[i];
            } else {
                const float* src = (const float*)Ap + (size_t)(gm + r) * DM + k0 + cb;
                #pragma unroll
                for (int i = 0; i < 16; ++i) lA[r * LDA + cb + i] = (__bf16)src[i];
            }
        }
        {   // B tile transposed: thread reads 8 contiguous W elements, scatters
            const int kk = tid >> 3;
            const int nb = (tid & 7) * 8;
            const float* src = W + (size_t)(k0 + kk) * DM + gn + nb;
            #pragma unroll
            for (int i = 0; i < 8; ++i) lBt[(nb + i) * LDB + kk] = (__bf16)src[i];
        }
        __syncthreads();

        // A fragment: 16x32, lane holds row (lrow), K = hf*8..+7 and +16..+23
        BF16x16 af;
        const int arow = wv * 16 + lrow;
        const int koff = hf * 8;
        af.h[0] = *(const v8bf*)&lA[arow * LDA + koff];
        af.h[1] = *(const v8bf*)&lA[arow * LDA + koff + 16];
        #pragma unroll
        for (int nt = 0; nt < 4; ++nt) {
            // B fragment: 32x16, lane = column, K = hf*8..+7 and +16..+23
            BF16x16 bf;
            const int bn  = nt * 16 + lrow;
            const int klo = hf * 8;
            bf.h[0] = *(const v8bf*)&lBt[bn * LDB + klo];
            bf.h[1] = *(const v8bf*)&lBt[bn * LDB + klo + 16];
            acc[nt] = wmma_bf16(af.v, bf.v, acc[nt]);
        }
    }

    // Epilogue: C layout = lane holds column lrow, rows v + 8*hf
    #pragma unroll
    for (int nt = 0; nt < 4; ++nt) {
        const int col = gn + nt * 16 + lrow;
        const float bv = bias[col];
        #pragma unroll
        for (int v = 0; v < 8; ++v) {
            const int m  = gm + wv * 16 + v + 8 * hf;
            const float val = acc[nt][v] + bv;
            if (OUT_SPLIT) {
                const int b = m / S_LEN, s = m % S_LEN;
                const int h = col / DH,  d = col % DH;
                ((__bf16*)Out)[(((size_t)(b * HEADS + h) * S_LEN) + s) * DH + d] =
                    (__bf16)val;
            } else {
                ((float*)Out)[(size_t)m * DM + col] = val;
            }
        }
    }
}

// ---------------------------------------------------------------------------
// Flash attention over [B,H,S,DH] bf16 Q/K/V with int mask and online softmax.
// Block = (bh, 128 q-rows); each wave owns 16 q rows x full DH=64.
// Scores: D = Qfrag(16x32) x K^T(32x16) via direct global B-frag loads.
// Context: D = Pfrag(16x32, via LDS transpose) x Vt(32x16, LDS-staged).
// ---------------------------------------------------------------------------
__global__ __launch_bounds__(256) void attn_kernel(
    const __bf16* __restrict__ Q, const __bf16* __restrict__ K,
    const __bf16* __restrict__ V, const int* __restrict__ mask,
    __bf16* __restrict__ ctx) {
    constexpr int LDV = 40;
    constexpr int LDP = 40;
    __shared__ __bf16 lVt[DH * LDV];        // V^T tile: [dim][key 0..31]
    __shared__ __bf16 lP[8 * 16 * LDP];     // per-wave P tile [qrow][key 0..31]

    const int tid  = threadIdx.x;
    const int wv   = tid >> 5;
    const int lane = tid & 31;
    const int lrow = lane & 15;
    const int hf   = lane >> 4;

    const int bh   = blockIdx.x;
    const int b    = bh / HEADS;
    const int h    = bh % HEADS;
    const int qrow0 = blockIdx.y * 128 + wv * 16;

    const __bf16* Qbh = Q + (size_t)bh * S_LEN * DH;
    const __bf16* Kbh = K + (size_t)bh * S_LEN * DH;
    const __bf16* Vbh = V + (size_t)bh * S_LEN * DH;

    // Q fragments: two 16x32 A-frags covering DH=64
    BF16x16 qf[2];
    {
        const __bf16* qp = Qbh + (size_t)(qrow0 + lrow) * DH;
        #pragma unroll
        for (int c = 0; c < 2; ++c) {
            const int kg = c * 32 + hf * 8;
            qf[c].h[0] = *(const v8bf*)(qp + kg);
            qf[c].h[1] = *(const v8bf*)(qp + kg + 16);
        }
    }

    float mr[8], ls[8];
    #pragma unroll
    for (int v = 0; v < 8; ++v) { mr[v] = -3.0e38f; ls[v] = 0.f; }
    v8f acc[4] = {};

    for (int j = 0; j < S_LEN; j += 32) {
        __syncthreads();
        {   // stage V^T: contiguous global reads, scattered LDS stores
            const int kk = tid >> 3;
            const int db = (tid & 7) * 8;
            const __bf16* src = Vbh + (size_t)(j + kk) * DH + db;
            #pragma unroll
            for (int i = 0; i < 8; ++i) lVt[(db + i) * LDV + kk] = src[i];
        }
        __syncthreads();

        // scores for two 16-key tiles; B-frag rows of K read straight from HBM
        v8f s[2];
        #pragma unroll
        for (int t = 0; t < 2; ++t) {
            v8f sc = {};
            const __bf16* kp = Kbh + (size_t)(j + t * 16 + lrow) * DH;
            #pragma unroll
            for (int c = 0; c < 2; ++c) {
                BF16x16 bf;
                const int kg = c * 32 + hf * 8;
                bf.h[0] = *(const v8bf*)(kp + kg);
                bf.h[1] = *(const v8bf*)(kp + kg + 16);
                sc = wmma_bf16(qf[c].v, bf.v, sc);
            }
            s[t] = sc;
        }

        // scale + mask (mask==0 -> -1e9, matching reference)
        #pragma unroll
        for (int t = 0; t < 2; ++t) {
            #pragma unroll
            for (int v = 0; v < 8; ++v) {
                const int row = qrow0 + v + 8 * hf;
                const int key = j + t * 16 + lrow;
                const int mv  = mask[((size_t)b * S_LEN + row) * S_LEN + key];
                s[t][v] = (mv == 0) ? -1.0e9f : s[t][v] * 0.125f;
            }
        }

        // online softmax: reductions across lane bits 0..3 (within each half)
        float mnew[8], alpha[8];
        #pragma unroll
        for (int v = 0; v < 8; ++v) {
            float mx = fmaxf(s[0][v], s[1][v]);
            #pragma unroll
            for (int d = 1; d < 16; d <<= 1) mx = fmaxf(mx, __shfl_xor(mx, d, 32));
            mnew[v]  = fmaxf(mr[v], mx);
            alpha[v] = __expf(mr[v] - mnew[v]);
            mr[v]    = mnew[v];
        }

        __bf16* myP = &lP[wv * 16 * LDP];
        #pragma unroll
        for (int v = 0; v < 8; ++v) {
            const float p0 = __expf(s[0][v] - mnew[v]);
            const float p1 = __expf(s[1][v] - mnew[v]);
            float rs = p0 + p1;
            #pragma unroll
            for (int d = 1; d < 16; d <<= 1) rs += __shfl_xor(rs, d, 32);
            ls[v] = ls[v] * alpha[v] + rs;
            myP[(v + 8 * hf) * LDP + lrow]      = (__bf16)p0;
            myP[(v + 8 * hf) * LDP + 16 + lrow] = (__bf16)p1;
            #pragma unroll
            for (int nt = 0; nt < 4; ++nt) acc[nt][v] *= alpha[v];
        }

        // context += P(16x32) x V(32x16 per d-chunk)
        BF16x16 pf;
        pf.h[0] = *(const v8bf*)&myP[lrow * LDP + hf * 8];
        pf.h[1] = *(const v8bf*)&myP[lrow * LDP + hf * 8 + 16];
        #pragma unroll
        for (int nt = 0; nt < 4; ++nt) {
            BF16x16 vf;
            const int dd = nt * 16 + lrow;
            vf.h[0] = *(const v8bf*)&lVt[dd * LDV + hf * 8];
            vf.h[1] = *(const v8bf*)&lVt[dd * LDV + hf * 8 + 16];
            acc[nt] = wmma_bf16(pf.v, vf.v, acc[nt]);
        }
    }

    // normalize and write context merged-heads: [B,S,DM] bf16
    #pragma unroll
    for (int nt = 0; nt < 4; ++nt) {
        #pragma unroll
        for (int v = 0; v < 8; ++v) {
            const int row = qrow0 + v + 8 * hf;
            const int d   = nt * 16 + lrow;
            const float val = acc[nt][v] / ls[v];
            ctx[((size_t)b * S_LEN + row) * DM + h * DH + d] = (__bf16)val;
        }
    }
}

extern "C" void kernel_launch(void* const* d_in, const int* in_sizes, int n_in,
                              void* d_out, int out_size, void* d_ws, size_t ws_size,
                              hipStream_t stream) {
    (void)in_sizes; (void)n_in; (void)out_size; (void)ws_size;
    const float* query = (const float*)d_in[0];
    const int*   mask  = (const int*)d_in[1];
    const float* Wq = (const float*)d_in[2];
    const float* bq = (const float*)d_in[3];
    const float* Wk = (const float*)d_in[4];
    const float* bk = (const float*)d_in[5];
    const float* Wv = (const float*)d_in[6];
    const float* bv = (const float*)d_in[7];
    const float* Wo = (const float*)d_in[8];
    const float* bo = (const float*)d_in[9];
    float* out = (float*)d_out;

    const size_t elems = (size_t)BATCH * S_LEN * DM;
    __bf16* Qws = (__bf16*)d_ws;
    __bf16* Kws = Qws + elems;
    __bf16* Vws = Kws + elems;
    __bf16* Cws = Vws + elems;

    dim3 gblk(DM / 64, (BATCH * S_LEN) / 128);
    gemm_bias_kernel<false, true><<<gblk, 256, 0, stream>>>(query, Wq, bq, Qws);
    gemm_bias_kernel<false, true><<<gblk, 256, 0, stream>>>(query, Wk, bk, Kws);
    gemm_bias_kernel<false, true><<<gblk, 256, 0, stream>>>(query, Wv, bv, Vws);

    dim3 gattn(BATCH * HEADS, S_LEN / 128);
    attn_kernel<<<gattn, 256, 0, stream>>>(Qws, Kws, Vws, mask, Cws);

    gemm_bias_kernel<true, false><<<gblk, 256, 0, stream>>>(Cws, Wo, bo, out);
}
